// StyleAttentionNet_18287970746386
// MI455X (gfx1250) — matrix operand order repaired and checked by reading, not compiled, not run on hardware
//
#include <hip/hip_runtime.h>
#include <stdint.h>

// ---------------- types / constants ----------------
typedef __attribute__((ext_vector_type(16))) __bf16 v16bf;
typedef __attribute__((ext_vector_type(8)))  __bf16 v8bf;
typedef __attribute__((ext_vector_type(8)))  float  v8f;

#define BATCH 4
#define CCH   512
#define NSP   4096      // 64*64 spatial
#define EPSN  1e-5f

#define FA_BI 32        // query rows per workgroup
#define FA_BJ 64        // keys per inner iteration

// ---------------- helpers ----------------
static __device__ __forceinline__ unsigned short f2bf(float f) {
    unsigned u = __float_as_uint(f);
    unsigned r = (u + 0x7FFFu + ((u >> 16) & 1u)) >> 16;   // RNE
    return (unsigned short)r;
}

// A-operand fragment: 16x32 bf16. Lane l holds row M=l&15.
// lanes 0-15: e0..7 = K k0+0..7,  e8..15 = K k0+16..23
// lanes16-31: e0..7 = K k0+8..15, e8..15 = K k0+24..31   (ISA 16-bit A layout)
static __device__ __forceinline__ v16bf load_fragA(const unsigned short* base,
                                                   int row0, int k0, int ld, int lane) {
    int r = lane & 15, sel = lane >> 4;
    const unsigned short* p = base + (size_t)(row0 + r) * ld + k0 + sel * 8;
    union { v16bf v; v8bf h[2]; } u;
    u.h[0] = *(const v8bf*)(p);
    u.h[1] = *(const v8bf*)(p + 16);
    return u.v;
}

// B-operand fragment: 32x16 bf16 loaded from a (Nout x K) row-major buffer
// (column n of B == row n of Bs). Lane l holds column N=l&15.
// lanes 0-15: e0..7 = K k0+0..7, e8..15 = K k0+16..23; lanes 16-31: e0..7 = K k0+8..15, e8..15 = K k0+24..31.
static __device__ __forceinline__ v16bf load_fragB(const unsigned short* base,
                                                   int row0, int k0, int ld, int lane) {
    int n = lane & 15, sel = lane >> 4;
    const unsigned short* p = base + (size_t)(row0 + n) * ld + k0 + sel * 8;
    union { v16bf v; v8bf h[2]; } u;
    u.h[0] = *(const v8bf*)(p);
    u.h[1] = *(const v8bf*)(p + 16);
    return u.v;
}

static __device__ __forceinline__ v8f wmma_bf16(v16bf a, v16bf b, v8f c) {
    return __builtin_amdgcn_wmma_f32_16x16x32_bf16(false, a, false, b,
                                                   (short)0, c, false, false);
}

// Async copy 16 bytes global -> LDS (no VGPR round trip, tracked by ASYNCcnt).
static __device__ __forceinline__ void async_g2l_b128(void* lds_dst, const void* gsrc) {
    unsigned lds = (unsigned)(uintptr_t)lds_dst;   // ISA: LDS addr = generic[31:0]
    asm volatile("global_load_async_to_lds_b128 %0, %1, off"
                 :: "v"(lds), "v"(gsrc) : "memory");
}
static __device__ __forceinline__ void wait_async0() {
    asm volatile("s_wait_asynccnt 0x0" ::: "memory");
}

// ---------------- kernel 1: per-(b,c) instance-norm stats ----------------
__global__ void stats_kernel(const float* __restrict__ x,
                             float* __restrict__ mean, float* __restrict__ rstd) {
    __shared__ float s1[256], s2[256];
    int bc = blockIdx.x;                         // b*C + c
    const float* p = x + (size_t)bc * NSP;
    float s = 0.f, ss = 0.f;
    for (int i = threadIdx.x; i < NSP; i += 256) { float v = p[i]; s += v; ss += v * v; }
    s1[threadIdx.x] = s; s2[threadIdx.x] = ss;
    __syncthreads();
    for (int off = 128; off > 0; off >>= 1) {
        if (threadIdx.x < (unsigned)off) {
            s1[threadIdx.x] += s1[threadIdx.x + off];
            s2[threadIdx.x] += s2[threadIdx.x + off];
        }
        __syncthreads();
    }
    if (threadIdx.x == 0) {
        float m = s1[0] / (float)NSP;
        float var = (s2[0] - (float)NSP * m * m) / (float)(NSP - 1);   // ddof=1
        mean[bc] = m;
        rstd[bc] = rsqrtf(var + EPSN);
    }
}

// ---------------- kernel 2: (optional norm) + transpose CxN f32 -> NxC bf16 ----
__global__ void transpose_norm_kernel(const float* __restrict__ x,
                                      const float* __restrict__ mean,
                                      const float* __restrict__ rstd,
                                      int use_norm, unsigned short* __restrict__ xT) {
    __shared__ unsigned short tile[32][33];
    int n0 = blockIdx.x * 32, c0 = blockIdx.y * 32, b = blockIdx.z;
    int tx = threadIdx.x, ty = threadIdx.y;
    for (int i = 0; i < 4; ++i) {
        int r = ty + i * 8;
        int ch = c0 + r;
        float v = x[((size_t)(b * CCH + ch)) * NSP + n0 + tx];
        if (use_norm) v = (v - mean[b * CCH + ch]) * rstd[b * CCH + ch];
        tile[r][tx] = f2bf(v);
    }
    __syncthreads();
    for (int i = 0; i < 4; ++i) {
        int r = ty + i * 8;
        xT[((size_t)(b * NSP + n0 + r)) * CCH + c0 + tx] = tile[tx][r];
    }
}

// ---------------- kernel 3: f32 -> bf16 (weights) ----------------
__global__ void cvt_bf16_kernel(const float* __restrict__ x,
                                unsigned short* __restrict__ y, int n) {
    int i = blockIdx.x * 256 + threadIdx.x;
    if (i < n) y[i] = f2bf(x[i]);
}

// ---------------- kernel 4: generic NT GEMM, bf16 in / f32 accumulate ---------
// D[m][n] = sum_k A[m][k]*Bs[n][k]  (+bias)(+residual); block tile 128x128,
// 8 waves as 4(M)x2(N); wave tile 32x64 = 2x4 wmma accumulators.
__global__ void __launch_bounds__(256)
gemm_nt_kernel(const unsigned short* __restrict__ A, long long strideA, int lda,
               const unsigned short* __restrict__ Bm, long long strideB, int ldb,
               const float* __restrict__ bias, int biasMode,   // 0 none, 1 col, 2 row
               unsigned short* __restrict__ Dbf, float* __restrict__ Df,
               long long strideD, int ldd,
               const float* __restrict__ residual, int K) {
    int b = blockIdx.z;
    const unsigned short* Ab = A  + (size_t)b * strideA;
    const unsigned short* Bb = Bm + (size_t)b * strideB;

    int wave = threadIdx.x >> 5, lane = threadIdx.x & 31;
    int wm = wave & 3, wn = wave >> 2;
    int m0 = blockIdx.x * 128 + wm * 32;
    int n0 = blockIdx.y * 128 + wn * 64;

    v8f zero = {};
    v8f acc[2][4];
    for (int i = 0; i < 2; ++i)
        for (int j = 0; j < 4; ++j) acc[i][j] = zero;

    for (int k = 0; k < K; k += 32) {
        v16bf af[2], bfr[4];
        for (int i = 0; i < 2; ++i) af[i]  = load_fragA(Ab, m0 + 16 * i, k, lda, lane);
        for (int j = 0; j < 4; ++j) bfr[j] = load_fragB(Bb, n0 + 16 * j, k, ldb, lane);
        for (int i = 0; i < 2; ++i)
            for (int j = 0; j < 4; ++j)
                acc[i][j] = wmma_bf16(af[i], bfr[j], acc[i][j]);
    }

    int col = lane & 15, sel = lane >> 4;
    for (int i = 0; i < 2; ++i)
        for (int j = 0; j < 4; ++j) {
            int nn = n0 + 16 * j + col;
            for (int r = 0; r < 8; ++r) {
                int mm = m0 + 16 * i + r + 8 * sel;
                float v = acc[i][j][r];
                if (biasMode == 1) v += bias[nn];
                else if (biasMode == 2) v += bias[mm];
                size_t idx = (size_t)b * (size_t)strideD + (size_t)mm * ldd + nn;
                if (residual) v += residual[idx];
                if (Df) Df[idx] = v;
                else    Dbf[idx] = f2bf(v);
            }
        }
}

// ---------------- kernel 5: flash attention --------------------------------
// energy = Qt x Kt^T, online softmax over keys, Ot = softmax x Vm^T.
// Qt,Kt: [B][N][C] bf16; Vm: [B][C][N] bf16; Ot: [B][N][C] bf16.
__global__ void __launch_bounds__(256)
flash_attn_kernel(const unsigned short* __restrict__ Qt,
                  const unsigned short* __restrict__ Kt,
                  const unsigned short* __restrict__ Vm,
                  unsigned short* __restrict__ Ot) {
    __shared__ unsigned short qtile[FA_BI * CCH];          // 32x512 bf16 = 32 KB
    __shared__ float          Sf[FA_BI][FA_BJ + 1];        // padded vs bank conflicts
    __shared__ unsigned short Pb[FA_BI][FA_BJ];            // bf16 probs (A-operand)
    __shared__ float mrow[FA_BI], lrow[FA_BI], arow[FA_BI];
    __shared__ float pmax[FA_BI][8], psum[FA_BI][8];

    int b  = blockIdx.y;
    int i0 = blockIdx.x * FA_BI;
    const unsigned short* Qb = Qt + (size_t)b * NSP * CCH;
    const unsigned short* Kb = Kt + (size_t)b * NSP * CCH;
    const unsigned short* Vb = Vm + (size_t)b * CCH * NSP;
    unsigned short*       Ob = Ot + (size_t)b * NSP * CCH;

    // ---- stage Q block into LDS with async global->LDS b128 copies ----
    {
        const unsigned short* src = Qb + (size_t)i0 * CCH;
        // 32x512 bf16 = 32KB = 2048 x 16B; 256 threads x 8 chunks
        for (int t = threadIdx.x; t < FA_BI * CCH / 8; t += 256)
            async_g2l_b128(&qtile[t * 8], src + (size_t)t * 8);
        wait_async0();
    }
    if (threadIdx.x < FA_BI) { mrow[threadIdx.x] = -3.0e38f; lrow[threadIdx.x] = 0.f; }
    __syncthreads();

    int wave = threadIdx.x >> 5, lane = threadIdx.x & 31;
    int col = lane & 15, sel = lane >> 4;
    int stm = wave & 1, stn = wave >> 1;   // S-tile mapping: 2(M) x 4(N)
    int c0  = wave * 64;                   // O mapping: this wave's channel slice
    int srow = threadIdx.x >> 3;           // softmax: 8 threads per query row
    int ssub = threadIdx.x & 7;

    v8f zero = {};
    v8f oacc[2][4];
    for (int i = 0; i < 2; ++i)
        for (int j = 0; j < 4; ++j) oacc[i][j] = zero;

    for (int j0 = 0; j0 < NSP; j0 += FA_BJ) {
        // warm L2->WGP path for the next iteration's K and V tiles
        if (j0 + FA_BJ < NSP) {
            __builtin_prefetch(Kb + (size_t)(j0 + FA_BJ + stn * 16 + col) * CCH + sel * 256, 0, 3);
            __builtin_prefetch(Vb + (size_t)(c0 + 16 * sel + col) * NSP + j0 + FA_BJ, 0, 3);
        }

        // ---- S tile: each wave computes one 16x16 tile of the 32x64 S block
        v8f sacc = zero;
        for (int k = 0; k < CCH; k += 32) {
            v16bf aq = load_fragA(qtile, stm * 16, k, CCH, lane);            // LDS
            v16bf bk = load_fragB(Kb, j0 + stn * 16, k, CCH, lane);          // L2
            sacc = wmma_bf16(aq, bk, sacc);
        }
        for (int r = 0; r < 8; ++r)
            Sf[stm * 16 + r + 8 * sel][stn * 16 + col] = sacc[r];
        __syncthreads();

        // ---- online softmax, 8 threads per query row (all 256 threads busy)
        float m_old = mrow[srow];
        {
            float mx = -3.0e38f;
            for (int jj = 0; jj < 8; ++jj) mx = fmaxf(mx, Sf[srow][ssub * 8 + jj]);
            pmax[srow][ssub] = mx;
        }
        __syncthreads();
        {
            float mx = m_old;
            for (int t = 0; t < 8; ++t) mx = fmaxf(mx, pmax[srow][t]);
            float s = 0.f;
            for (int jj = 0; jj < 8; ++jj) {
                float p = __expf(Sf[srow][ssub * 8 + jj] - mx);
                s += p;
                Pb[srow][ssub * 8 + jj] = f2bf(p);
            }
            psum[srow][ssub] = s;
            if (ssub == 0) {
                float alpha = __expf(m_old - mx);
                arow[srow] = alpha;
                mrow[srow] = mx;
            }
        }
        __syncthreads();
        if (ssub == 0) {
            float tot = 0.f;
            for (int t = 0; t < 8; ++t) tot += psum[srow][t];
            lrow[srow] = lrow[srow] * arow[srow] + tot;
        }

        // ---- rescale O and accumulate P x V^T (wave: 32 rows x 64 channels)
        for (int i = 0; i < 2; ++i)
            for (int r = 0; r < 8; ++r) {
                float a = arow[i * 16 + r + 8 * sel];
                for (int j = 0; j < 4; ++j) oacc[i][j][r] *= a;
            }
        for (int kj = 0; kj < FA_BJ; kj += 32) {
            v16bf ap[2];
            ap[0] = load_fragA((const unsigned short*)Pb, 0,  kj, FA_BJ, lane);
            ap[1] = load_fragA((const unsigned short*)Pb, 16, kj, FA_BJ, lane);
            for (int j = 0; j < 4; ++j) {
                v16bf bv = load_fragB(Vb, c0 + 16 * j, j0 + kj, NSP, lane);
                for (int i = 0; i < 2; ++i)
                    oacc[i][j] = wmma_bf16(ap[i], bv, oacc[i][j]);
            }
        }
        __syncthreads();
    }

    // ---- normalize by l and store Ot[i][c] as bf16
    for (int i = 0; i < 2; ++i)
        for (int r = 0; r < 8; ++r) {
            int row = i * 16 + r + 8 * sel;
            float rl = 1.f / lrow[row];
            for (int j = 0; j < 4; ++j) {
                int cc = c0 + 16 * j + col;
                Ob[(size_t)(i0 + row) * CCH + cc] = f2bf(oacc[i][j][r] * rl);
            }
        }
}

// ---------------- host side ----------------
extern "C" void kernel_launch(void* const* d_in, const int* in_sizes, int n_in,
                              void* d_out, int out_size, void* d_ws, size_t ws_size,
                              hipStream_t stream) {
    (void)in_sizes; (void)n_in; (void)out_size; (void)ws_size;
    const float* content = (const float*)d_in[0];
    const float* style   = (const float*)d_in[1];
    const float* Wq = (const float*)d_in[2];  const float* bq = (const float*)d_in[3];
    const float* Wk = (const float*)d_in[4];  const float* bk = (const float*)d_in[5];
    const float* Wv = (const float*)d_in[6];  const float* bv = (const float*)d_in[7];
    const float* Wo = (const float*)d_in[8];  const float* bo = (const float*)d_in[9];
    float* out = (float*)d_out;

    // workspace carve-up (256B aligned)
    char* base = (char*)d_ws;
    size_t off = 0;
    auto alloc = [&](size_t bytes) { char* p = base + off; off += (bytes + 255) & ~(size_t)255; return p; };
    const size_t actElems = (size_t)BATCH * NSP * CCH;       // 8M elems
    float* meanC = (float*)alloc(BATCH * CCH * 4);
    float* rstdC = (float*)alloc(BATCH * CCH * 4);
    float* meanS = (float*)alloc(BATCH * CCH * 4);
    float* rstdS = (float*)alloc(BATCH * CCH * 4);
    unsigned short* Wq16 = (unsigned short*)alloc((size_t)CCH * CCH * 2);
    unsigned short* Wk16 = (unsigned short*)alloc((size_t)CCH * CCH * 2);
    unsigned short* Wv16 = (unsigned short*)alloc((size_t)CCH * CCH * 2);
    unsigned short* Wo16 = (unsigned short*)alloc((size_t)CCH * CCH * 2);
    unsigned short* XcT  = (unsigned short*)alloc(actElems * 2);   // mvn(content)^T
    unsigned short* XsT  = (unsigned short*)alloc(actElems * 2);   // mvn(style)^T
    unsigned short* SsT  = (unsigned short*)alloc(actElems * 2);   // style^T (raw)
    unsigned short* Qt   = (unsigned short*)alloc(actElems * 2);   // [B][N][C]
    unsigned short* Kt   = (unsigned short*)alloc(actElems * 2);   // [B][N][C]
    unsigned short* Vmt  = (unsigned short*)alloc(actElems * 2);   // [B][C][N]
    unsigned short* Ot   = (unsigned short*)alloc(actElems * 2);   // [B][N][C]

    const long long sAct = (long long)NSP * CCH;    // per-batch activation stride
    const int nW = CCH * CCH;

    // 1) instance-norm statistics
    stats_kernel<<<BATCH * CCH, 256, 0, stream>>>(content, meanC, rstdC);
    stats_kernel<<<BATCH * CCH, 256, 0, stream>>>(style,   meanS, rstdS);

    // 2) weights -> bf16
    cvt_bf16_kernel<<<(nW + 255) / 256, 256, 0, stream>>>(Wq, Wq16, nW);
    cvt_bf16_kernel<<<(nW + 255) / 256, 256, 0, stream>>>(Wk, Wk16, nW);
    cvt_bf16_kernel<<<(nW + 255) / 256, 256, 0, stream>>>(Wv, Wv16, nW);
    cvt_bf16_kernel<<<(nW + 255) / 256, 256, 0, stream>>>(Wo, Wo16, nW);

    // 3) normalize + transpose to (N, C) bf16
    dim3 tb(32, 8), tg(NSP / 32, CCH / 32, BATCH);
    transpose_norm_kernel<<<tg, tb, 0, stream>>>(content, meanC, rstdC, 1, XcT);
    transpose_norm_kernel<<<tg, tb, 0, stream>>>(style,   meanS, rstdS, 1, XsT);
    transpose_norm_kernel<<<tg, tb, 0, stream>>>(style,   meanS, rstdS, 0, SsT);

    // 4) Qt = XcT x Wq^T + bq(col) ; Kt = XsT x Wk^T + bk(col)   -> [B][N][C]
    dim3 g1(NSP / 128, CCH / 128, BATCH);
    gemm_nt_kernel<<<g1, 256, 0, stream>>>(XcT, sAct, CCH, Wq16, 0, CCH,
                                           bq, 1, Qt, nullptr, sAct, CCH, nullptr, CCH);
    gemm_nt_kernel<<<g1, 256, 0, stream>>>(XsT, sAct, CCH, Wk16, 0, CCH,
                                           bk, 1, Kt, nullptr, sAct, CCH, nullptr, CCH);

    // 5) Vm = Wv x style^T^T + bv(row)  -> [B][C][N]
    dim3 g2(CCH / 128, NSP / 128, BATCH);
    gemm_nt_kernel<<<g2, 256, 0, stream>>>(Wv16, 0, CCH, SsT, sAct, CCH,
                                           bv, 2, Vmt, nullptr, sAct, NSP, nullptr, CCH);

    // 6) fused attention: softmax(Qt Kt^T) Vm^T -> Ot [B][N][C]
    dim3 g3(NSP / FA_BI, BATCH);
    flash_attn_kernel<<<g3, 256, 0, stream>>>(Qt, Kt, Vmt, Ot);

    // 7) out = Wo x Ot^T + bo(row) + content  -> f32 [B][C][H][W]
    gemm_nt_kernel<<<g2, 256, 0, stream>>>(Wo16, 0, CCH, Ot, sAct, CCH,
                                           bo, 2, nullptr, out, sAct, NSP, content, CCH);
}
